// EncoderLayer_3977139716822
// MI455X (gfx1250) — matrix operand order repaired and checked by reading, not compiled, not run on hardware
//
#include <hip/hip_runtime.h>
#include <math.h>
#include <stdint.h>

// ---------------------------------------------------------------------------
// CDNA5 (gfx1250) transformer encoder layer, bf16 WMMA + f32 accumulate.
// Weight GEMMs use the Tensor Data Mover (tensor_load_to_lds + s_wait_tensorcnt)
// to stage shared B tiles into LDS, double-buffered, 8 waves per workgroup.
//   out  = LN2( LN1(attn_out + x) -> FFN -> +res )   [2,2048,1024] f32
//   attn = softmax(Q K^T / 8)                        [2,16,2048,2048] f32
// d_out = out || attn (flat, f32).
// ---------------------------------------------------------------------------

typedef __bf16 bf16;
typedef __attribute__((ext_vector_type(16))) __bf16 v16bf;
typedef __attribute__((ext_vector_type(8)))  __bf16 v8bf;
typedef __attribute__((ext_vector_type(8)))  float  v8f;
typedef __attribute__((ext_vector_type(4)))  unsigned int u32x4;
typedef __attribute__((ext_vector_type(8)))  int i32x8;
typedef __attribute__((ext_vector_type(4)))  int i32x4;

#define SEQ     2048
#define NBATCH  2
#define NHEAD   16
#define DK      64
#define DMODEL  1024
#define DINNER  4096
#define TOK     (NBATCH*SEQ)     // 4096 tokens
#define BHH     (NBATCH*NHEAD)   // 32 (batch*heads)

#if defined(__AMDGCN__) && __has_builtin(__builtin_amdgcn_tensor_load_to_lds) && \
    __has_builtin(__builtin_amdgcn_s_wait_tensorcnt)
#define HAVE_TDM 1
#else
#define HAVE_TDM 0
#endif

// ---- WMMA bf16 fragment loaders (contiguous k = 16*hi + s, not the ISA table order; A and B match) ----
// A 16x32 (16-bit): lanes 0-15 -> M=lane, K=0..15; lanes 16-31 -> K=16..31.
__device__ __forceinline__ v16bf load_a_bf16(const bf16* __restrict__ row, int k0, int hi) {
  return *(const v16bf*)(row + k0 + hi * 16);
}
// B 32x16 (16-bit): lanes 0-15 -> N=lane, K=0..15; lanes 16-31 -> K=16..31.
__device__ __forceinline__ v16bf load_b_bf16(const bf16* __restrict__ row, int k0, int hi) {
  return *(const v16bf*)(row + k0 + hi * 16);
}
// Same A layout, converting f32 source (attention probs) to bf16 on the fly.
__device__ __forceinline__ v16bf load_a_f32(const float* __restrict__ row, int k0, int hi) {
  union { v16bf v; bf16 e[16]; } u;
  const float* p0 = row + k0 + hi * 16;
  const float* p1 = row + k0 + hi * 16 + 8;
#pragma unroll
  for (int i = 0; i < 8; ++i) { u.e[i] = (bf16)p0[i]; u.e[8 + i] = (bf16)p1[i]; }
  return u.v;
}

__device__ __forceinline__ v8f wmma_bf16(v16bf a, v16bf b, v8f c) {
  // 8 args: (neg_a, A, neg_b, B, c_mod, C, reuse_a, reuse_b)
  return __builtin_amdgcn_wmma_f32_16x16x32_bf16(false, a, false, b, (short)0, c,
                                                 false, false);
}

enum { EPI_QK = 0, EPI_VT = 1, EPI_F32_RES = 2, EPI_GELU = 3, EPI_SCORES = 4, EPI_CTX = 5 };

// ---------------------------------------------------------------------------
// Shared epilogue. C/D layout: lane g = N column; VGPR r -> M = r + 8*hi.
// ---------------------------------------------------------------------------
template <int EPI>
__device__ __forceinline__ void epilogue(const v8f (&acc)[4][4], int m0, int n0,
                                         int g, int hi, int N, int z,
                                         const float* __restrict__ bias,
                                         const float* __restrict__ resid,
                                         char* __restrict__ outp) {
#pragma unroll
  for (int j = 0; j < 4; ++j) {
    const int nn = n0 + j * 16 + g;
    const float bv = bias ? bias[nn] : 0.0f;
#pragma unroll
    for (int i = 0; i < 4; ++i) {
#pragma unroll
      for (int r = 0; r < 8; ++r) {
        const int mm = m0 + i * 16 + hi * 8 + r;
        float v = acc[i][j][r] + bv;
        if (EPI == EPI_QK) {
          // q/k: [B,H,S,DK] bf16;  mm = token, nn = h*64+d
          int b = mm >> 11, s = mm & (SEQ - 1), h = nn >> 6, d = nn & (DK - 1);
          ((bf16*)outp)[(((size_t)(b * NHEAD + h) * SEQ + s) << 6) + d] = (bf16)v;
        } else if (EPI == EPI_VT) {
          // v transposed: [B,H,DK,S] bf16 (K-contiguous for ctx GEMM B operand)
          int b = mm >> 11, s = mm & (SEQ - 1), h = nn >> 6, d = nn & (DK - 1);
          ((bf16*)outp)[(((size_t)(b * NHEAD + h) * DK + d) << 11) + s] = (bf16)v;
        } else if (EPI == EPI_F32_RES) {
          size_t o = (size_t)mm * (size_t)N + nn;
          ((float*)outp)[o] = v + (resid ? resid[o] : 0.f);
        } else if (EPI == EPI_GELU) {
          float ge = 0.5f * v * (1.0f + erff(v * 0.70710678118654752f));
          ((bf16*)outp)[(size_t)mm * (size_t)N + nn] = (bf16)ge;
        } else if (EPI == EPI_SCORES) {
          ((float*)outp)[(size_t)mm * SEQ + nn] = v * 0.125f;  // 1/sqrt(64)
        } else {  // EPI_CTX: ctx[b, s, h*64+d] bf16 (token-major for Wo GEMM)
          int b = z >> 4, h = z & 15;
          ((bf16*)outp)[((size_t)(b * SEQ + mm) * DMODEL) + h * DK + nn] = (bf16)v;
        }
      }
    }
  }
}

#if HAVE_TDM
// Build a 2D TDM descriptor (CDNA5 ISA) and issue
// tensor_load_to_lds: tile = 64 rows x 64 bf16, row stride ldb elements.
// This toolchain's builtin takes 6 args:
//   (uint32x4 g0, int32x8 g1, int32x4 g2, int32x4 g3, int32x8 g4, i32 cpol)
__device__ __forceinline__ void tdm_stage(const bf16* gsrc, unsigned ldsoff,
                                          int ldb, int K, int N) {
  unsigned long long ga = (unsigned long long)(uintptr_t)gsrc;
  u32x4 g0;
  g0[0] = 1u;                                   // count=1, user descriptor
  g0[1] = ldsoff;                               // lds_addr (bytes)
  g0[2] = (unsigned)(ga & 0xffffffffu);         // global_addr[31:0]
  g0[3] = (unsigned)((ga >> 32) & 0x1ffffffu) | (2u << 30);  // addr[56:32]|type=2
  i32x8 g1;
  g1[0] = 1 << 16;                              // data_size=1 (2 bytes), mask=0
  g1[1] = (int)(((unsigned)K & 0xffffu) << 16); // tensor_dim0[15:0]
  g1[2] = (int)(((unsigned)K >> 16) | (((unsigned)N & 0xffffu) << 16));
  g1[3] = (int)(((unsigned)N >> 16) | (64u << 16));  // dim1 hi | tile_dim0=64
  g1[4] = 64;                                   // tile_dim1=64, tile_dim2=0
  g1[5] = (int)(unsigned)ldb;                   // tensor_dim0_stride[31:0]
  g1[6] = 0;
  g1[7] = 0;
  i32x4 z4 = {0, 0, 0, 0};
  i32x8 z8 = {0, 0, 0, 0, 0, 0, 0, 0};
  __builtin_amdgcn_tensor_load_to_lds(g0, g1, z4, z4, z8, 0);
}
#endif

// ---------------------------------------------------------------------------
// Workgroup WMMA GEMM with TDM-staged B tile:  C = A[M,K] * W[N,K]^T (+epi).
// 256 threads = 8 waves; WG tile 512(M) x 64(N); K chunked by 64; B tile
// (64 rows x 64 bf16 = 8KB) staged to LDS by wave 0 via the Tensor Data Mover,
// double-buffered so the DMA of chunk ck+1 overlaps WMMAs on chunk ck.
// ---------------------------------------------------------------------------
template <int EPI>
__global__ void __launch_bounds__(256)
gemm_wmma_tdm_k(const bf16* __restrict__ A, int lda,
                const bf16* __restrict__ Bw, int ldb,
                const float* __restrict__ bias,
                const float* __restrict__ resid,
                void* __restrict__ outv, int N, int K) {
  __shared__ __align__(32) unsigned short smemB[2][64 * 64];
  const int tid  = threadIdx.x;
  const int w    = tid >> 5;
  const int lane = tid & 31;
  const int g = lane & 15, hi = lane >> 4;
  const int n0 = blockIdx.x * 64;
  const int m0 = blockIdx.y * 512 + w * 64;

  const bf16* ar[4];
#pragma unroll
  for (int i = 0; i < 4; ++i) ar[i] = A + (size_t)(m0 + i * 16 + g) * (size_t)lda;

  const v8f vzero = {0.f, 0.f, 0.f, 0.f, 0.f, 0.f, 0.f, 0.f};
  v8f acc[4][4];
#pragma unroll
  for (int i = 0; i < 4; ++i)
#pragma unroll
    for (int j = 0; j < 4; ++j) acc[i][j] = vzero;

  const int nck = K >> 6;
#if HAVE_TDM
  if (w == 0)
    tdm_stage(Bw + (size_t)n0 * ldb,
              (unsigned)(unsigned long long)(uintptr_t)(void*)&smemB[0][0],
              ldb, K, N);
  for (int ck = 0; ck < nck; ++ck) {
    const int buf = ck & 1;
    if (w == 0) __builtin_amdgcn_s_wait_tensorcnt(0);
    __syncthreads();
    if (w == 0 && ck + 1 < nck)
      tdm_stage(Bw + (size_t)n0 * ldb + (size_t)(ck + 1) * 64,
                (unsigned)(unsigned long long)(uintptr_t)(void*)&smemB[buf ^ 1][0],
                ldb, K, N);
#else
  for (int ck = 0; ck < nck; ++ck) {
    const int buf = 0;
    __syncthreads();
    {  // cooperative fallback copy: 256 threads x 32B
      const int r = tid >> 2, c = (tid & 3) * 16;
      *(v16bf*)&smemB[0][r * 64 + c] =
          *(const v16bf*)(Bw + (size_t)(n0 + r) * ldb + (size_t)ck * 64 + c);
    }
    __syncthreads();
#endif
#pragma unroll
    for (int t = 0; t < 2; ++t) {
      const int kk = t * 32;
      const int kg = ck * 64 + kk;
      v16bf af[4], bfm[4];
#pragma unroll
      for (int i = 0; i < 4; ++i) af[i] = load_a_bf16(ar[i], kg, hi);
#pragma unroll
      for (int j = 0; j < 4; ++j) {
        // B rows live packed in LDS: row r at r*64 elements.
        const unsigned short* lrow = &smemB[buf][(j * 16 + g) * 64 + kk + hi * 16];
        bfm[j] = *(const v16bf*)lrow;
      }
#pragma unroll
      for (int i = 0; i < 4; ++i)
#pragma unroll
        for (int j = 0; j < 4; ++j) acc[i][j] = wmma_bf16(af[i], bfm[j], acc[i][j]);
    }
#if HAVE_TDM
    __syncthreads();
#endif
  }
  epilogue<EPI>(acc, m0, n0, g, hi, N, 0, bias, resid, (char*)outv);
}

// ---------------------------------------------------------------------------
// Per-wave WMMA GEMM (attention: scores and ctx), batched over blockIdx.z.
// ---------------------------------------------------------------------------
template <int EPI, bool AF32>
__global__ void __launch_bounds__(32)
gemm_wmma_k(const void* __restrict__ Av, int lda,
            const bf16* __restrict__ Bw, int ldb,
            const float* __restrict__ bias,
            const float* __restrict__ resid,
            void* __restrict__ outv,
            int N, int K,
            size_t zsa, size_t zsb, size_t zOutBytes) {
  const int z    = blockIdx.z;
  const int n0   = blockIdx.x * 64;
  const int m0   = blockIdx.y * 64;
  const int lane = threadIdx.x;
  const int g    = lane & 15;
  const int hi   = lane >> 4;

  const bf16*  Ab   = (const bf16*)Av  + (AF32 ? 0 : zsa * (size_t)z);
  const float* Af   = (const float*)Av + (AF32 ? zsa * (size_t)z : 0);
  const bf16*  Bz   = Bw + zsb * (size_t)z;
  char*        outp = (char*)outv + zOutBytes * (size_t)z;

  const bf16*  arb[4];
  const float* arf[4];
  const bf16*  br[4];
#pragma unroll
  for (int i = 0; i < 4; ++i) {
    size_t mrow = (size_t)(m0 + i * 16 + g) * (size_t)lda;
    arb[i] = Ab + mrow;
    arf[i] = Af + mrow;
    br[i]  = Bz + (size_t)(n0 + i * 16 + g) * (size_t)ldb;
  }

  const v8f vzero = {0.f, 0.f, 0.f, 0.f, 0.f, 0.f, 0.f, 0.f};
  v8f acc[4][4];
#pragma unroll
  for (int i = 0; i < 4; ++i)
#pragma unroll
    for (int j = 0; j < 4; ++j) acc[i][j] = vzero;

  for (int k0 = 0; k0 < K; k0 += 32) {
    v16bf af[4], bfm[4];
#pragma unroll
    for (int i = 0; i < 4; ++i)
      af[i] = AF32 ? load_a_f32(arf[i], k0, hi) : load_a_bf16(arb[i], k0, hi);
#pragma unroll
    for (int j = 0; j < 4; ++j) bfm[j] = load_b_bf16(br[j], k0, hi);
#pragma unroll
    for (int i = 0; i < 4; ++i)
#pragma unroll
      for (int j = 0; j < 4; ++j) acc[i][j] = wmma_bf16(af[i], bfm[j], acc[i][j]);
  }
  epilogue<EPI>(acc, m0, n0, g, hi, N, z, bias, resid, outp);
}

// ---------------------------------------------------------------------------
// Row softmax over 2048 columns, in place (f32 attn region of d_out).
// ---------------------------------------------------------------------------
__global__ void __launch_bounds__(256) softmax_rows_k(float* __restrict__ attn) {
  __shared__ float red[256];
  const int tid = threadIdx.x;
  float* p = attn + (size_t)blockIdx.x * SEQ;
  float v[8];
#pragma unroll
  for (int i = 0; i < 8; ++i) v[i] = p[tid + i * 256];
  float m = v[0];
#pragma unroll
  for (int i = 1; i < 8; ++i) m = fmaxf(m, v[i]);
  red[tid] = m; __syncthreads();
  for (int off = 128; off > 0; off >>= 1) {
    if (tid < off) red[tid] = fmaxf(red[tid], red[tid + off]);
    __syncthreads();
  }
  m = red[0]; __syncthreads();
  float s = 0.f;
#pragma unroll
  for (int i = 0; i < 8; ++i) { v[i] = expf(v[i] - m); s += v[i]; }
  red[tid] = s; __syncthreads();
  for (int off = 128; off > 0; off >>= 1) {
    if (tid < off) red[tid] += red[tid + off];
    __syncthreads();
  }
  const float inv = 1.0f / red[0];
#pragma unroll
  for (int i = 0; i < 8; ++i) p[tid + i * 256] = v[i] * inv;
}

// ---------------------------------------------------------------------------
// LayerNorm over D_MODEL=1024; writes f32 always, bf16 copy optionally.
// ---------------------------------------------------------------------------
__global__ void __launch_bounds__(256)
layernorm_k(const float* __restrict__ in, const float* __restrict__ gam,
            const float* __restrict__ bet, float* __restrict__ outf,
            bf16* __restrict__ outb) {
  __shared__ float red[256];
  const int tid = threadIdx.x;
  const float* x = in + (size_t)blockIdx.x * DMODEL;
  float v[4];
#pragma unroll
  for (int i = 0; i < 4; ++i) v[i] = x[tid + i * 256];
  red[tid] = v[0] + v[1] + v[2] + v[3]; __syncthreads();
  for (int off = 128; off > 0; off >>= 1) {
    if (tid < off) red[tid] += red[tid + off];
    __syncthreads();
  }
  const float mu = red[0] * (1.0f / DMODEL); __syncthreads();
  float d2 = 0.f;
#pragma unroll
  for (int i = 0; i < 4; ++i) { float d = v[i] - mu; d2 += d * d; }
  red[tid] = d2; __syncthreads();
  for (int off = 128; off > 0; off >>= 1) {
    if (tid < off) red[tid] += red[tid + off];
    __syncthreads();
  }
  const float inv = rsqrtf(red[0] * (1.0f / DMODEL) + 1e-6f);
#pragma unroll
  for (int i = 0; i < 4; ++i) {
    const int c = tid + i * 256;
    const float y = (v[i] - mu) * inv * gam[c] + bet[c];
    outf[(size_t)blockIdx.x * DMODEL + c] = y;
    if (outb) outb[(size_t)blockIdx.x * DMODEL + c] = (bf16)y;
  }
}

__global__ void __launch_bounds__(256)
cvt_f32_bf16_k(const float* __restrict__ in, bf16* __restrict__ out, int n) {
  const int i = blockIdx.x * 256 + threadIdx.x;
  if (i < n) out[i] = (bf16)in[i];
}

// ---------------------------------------------------------------------------
extern "C" void kernel_launch(void* const* d_in, const int* in_sizes, int n_in,
                              void* d_out, int out_size, void* d_ws, size_t ws_size,
                              hipStream_t stream) {
  (void)in_sizes; (void)n_in; (void)out_size; (void)ws_size;
  const float* enc  = (const float*)d_in[0];
  const float* wq   = (const float*)d_in[1];
  const float* bq   = (const float*)d_in[2];
  const float* wk   = (const float*)d_in[3];
  const float* bk   = (const float*)d_in[4];
  const float* wv   = (const float*)d_in[5];
  const float* bv   = (const float*)d_in[6];
  const float* wo   = (const float*)d_in[7];
  const float* bo   = (const float*)d_in[8];
  const float* ln1g = (const float*)d_in[9];
  const float* ln1b = (const float*)d_in[10];
  const float* w1   = (const float*)d_in[11];
  const float* b1   = (const float*)d_in[12];
  const float* w2   = (const float*)d_in[13];
  const float* b2   = (const float*)d_in[14];
  const float* ln2g = (const float*)d_in[15];
  const float* ln2b = (const float*)d_in[16];

  // Workspace carve-up (~152 MB total, 256B aligned).
  char* wp = (char*)d_ws;
  auto alloc = [&](size_t bytes) { char* p = wp; wp += (bytes + 255) & ~(size_t)255; return p; };
  bf16*  xb   = (bf16*)alloc((size_t)TOK * DMODEL * 2);      // enc bf16
  bf16*  wqb  = (bf16*)alloc((size_t)DMODEL * DMODEL * 2);
  bf16*  wkb  = (bf16*)alloc((size_t)DMODEL * DMODEL * 2);
  bf16*  wvb  = (bf16*)alloc((size_t)DMODEL * DMODEL * 2);
  bf16*  wob  = (bf16*)alloc((size_t)DMODEL * DMODEL * 2);
  bf16*  w1b  = (bf16*)alloc((size_t)DINNER * DMODEL * 2);
  bf16*  w2b  = (bf16*)alloc((size_t)DMODEL * DINNER * 2);
  bf16*  qb   = (bf16*)alloc((size_t)BHH * SEQ * DK * 2);    // [B,H,S,64]
  bf16*  kb   = (bf16*)alloc((size_t)BHH * SEQ * DK * 2);    // [B,H,S,64]
  bf16*  vtb  = (bf16*)alloc((size_t)BHH * SEQ * DK * 2);    // [B,H,64,S]
  bf16*  ctxb = (bf16*)alloc((size_t)TOK * DMODEL * 2);      // [T,1024]
  float* y1   = (float*)alloc((size_t)TOK * DMODEL * 4);     // ctx@Wo^T+bo+res
  float* xf   = (float*)alloc((size_t)TOK * DMODEL * 4);     // LN1 out f32
  bf16*  xb2  = (bf16*)alloc((size_t)TOK * DMODEL * 2);      // LN1 out bf16
  bf16*  hb   = (bf16*)alloc((size_t)TOK * DINNER * 2);      // gelu(FFN1)
  float* y2   = (float*)alloc((size_t)TOK * DMODEL * 4);     // FFN2 + res

  float* outO = (float*)d_out;
  float* attn = outO + (size_t)TOK * DMODEL;  // [B,H,S,S] f32 region of d_out

  auto cvt = [&](const float* s, bf16* d, int n) {
    cvt_f32_bf16_k<<<dim3((n + 255) / 256), dim3(256), 0, stream>>>(s, d, n);
  };
  cvt(enc, xb, TOK * DMODEL);
  cvt(wq, wqb, DMODEL * DMODEL);
  cvt(wk, wkb, DMODEL * DMODEL);
  cvt(wv, wvb, DMODEL * DMODEL);
  cvt(wo, wob, DMODEL * DMODEL);
  cvt(w1, w1b, DINNER * DMODEL);
  cvt(w2, w2b, DMODEL * DINNER);

  // --- QKV projections (M=4096, N=1024, K=1024), TDM-staged weights ---
  dim3 gQKV(DMODEL / 64, TOK / 512, 1);
  gemm_wmma_tdm_k<EPI_QK><<<gQKV, 256, 0, stream>>>(
      xb, DMODEL, wqb, DMODEL, bq, nullptr, qb, DMODEL, DMODEL);
  gemm_wmma_tdm_k<EPI_QK><<<gQKV, 256, 0, stream>>>(
      xb, DMODEL, wkb, DMODEL, bk, nullptr, kb, DMODEL, DMODEL);
  gemm_wmma_tdm_k<EPI_VT><<<gQKV, 256, 0, stream>>>(
      xb, DMODEL, wvb, DMODEL, bv, nullptr, vtb, DMODEL, DMODEL);

  // --- scores = (Q K^T)/8 -> attn region of d_out (per b,h; M=N=2048, K=64) ---
  gemm_wmma_k<EPI_SCORES, false><<<dim3(SEQ / 64, SEQ / 64, BHH), 32, 0, stream>>>(
      qb, DK, kb, DK, nullptr, nullptr, attn, SEQ, DK,
      (size_t)SEQ * DK, (size_t)SEQ * DK, (size_t)SEQ * SEQ * 4);

  // --- softmax in place over attn rows ---
  softmax_rows_k<<<dim3(BHH * SEQ), 256, 0, stream>>>(attn);

  // --- ctx = attn @ V  (A = f32 attn converted on load; B = vT; M=2048,N=64,K=2048)
  gemm_wmma_k<EPI_CTX, true><<<dim3(1, SEQ / 64, BHH), 32, 0, stream>>>(
      attn, SEQ, vtb, SEQ, nullptr, nullptr, ctxb, DK, SEQ,
      (size_t)SEQ * SEQ, (size_t)DK * SEQ, 0);

  // --- attention output projection + residual ---
  gemm_wmma_tdm_k<EPI_F32_RES><<<dim3(DMODEL / 64, TOK / 512, 1), 256, 0, stream>>>(
      ctxb, DMODEL, wob, DMODEL, bo, enc, y1, DMODEL, DMODEL);

  layernorm_k<<<dim3(TOK), 256, 0, stream>>>(y1, ln1g, ln1b, xf, xb2);

  // --- FFN1: h = gelu(x @ W1^T + b1)  (M=4096, N=4096, K=1024) ---
  gemm_wmma_tdm_k<EPI_GELU><<<dim3(DINNER / 64, TOK / 512, 1), 256, 0, stream>>>(
      xb2, DMODEL, w1b, DMODEL, b1, nullptr, hb, DINNER, DMODEL);

  // --- FFN2: y2 = h @ W2^T + b2 + x  (M=4096, N=1024, K=4096) ---
  gemm_wmma_tdm_k<EPI_F32_RES><<<dim3(DMODEL / 64, TOK / 512, 1), 256, 0, stream>>>(
      hb, DINNER, w2b, DINNER, b2, xf, y2, DMODEL, DINNER);

  layernorm_k<<<dim3(TOK), 256, 0, stream>>>(y2, ln2g, ln2b, outO, nullptr);
}
